// EGCL_19043884990716
// MI455X (gfx1250) — matrix operand order repaired and checked by reading, not compiled, not run on hardware
//
#include <hip/hip_runtime.h>
#include <hip/hip_bf16.h>

// ---------------- problem constants (match reference) ----------------
#define NNODE 2048
#define NEDGE 65536
#define FDIM  128
#define HDIM  256
#define NADIM 8
#define EADIM 4
#define DIN   261          // 2F + 1 + EA
#define KE    288          // DIN padded to multiple of 32
#define KNODE 416          // (F + H + NA)=392 padded to multiple of 32

typedef __attribute__((ext_vector_type(16))) _Float16 v16h;
typedef __attribute__((ext_vector_type(8)))  _Float16 v8h;
typedef __attribute__((ext_vector_type(8)))  float    v8f;

__device__ __forceinline__ float silu_f(float x) { return x / (1.f + __expf(-x)); }

// ---------------------------------------------------------------------
// Weight conversion: f32 [K, Nn] row-major  ->  f16 fragment layout.
// Fragment layout: half index = (((ntile*KB + kb)*32 + lane)*16 + i)
//   lane in [0,32): n = ntile*16 + (lane&15), k = kb*32 + (lane>>4)*8 + (i&7) + (i>>3)*16
// so each lane's 16 halves (32 bytes) are contiguous -> two b128 loads.
// Rows k >= K are zero padding.
// ---------------------------------------------------------------------
__global__ void egcl_cvt_weight_frag(const float* __restrict__ W,
                                     _Float16* __restrict__ Wo,
                                     int K, int Nn, int Kpad) {
    int o = blockIdx.x * blockDim.x + threadIdx.x;
    int total = Kpad * Nn;
    if (o >= total) return;
    int KB   = Kpad >> 5;
    int i    = o & 15;
    int lane = (o >> 4) & 31;
    int rest = o >> 9;
    int kb   = rest % KB;
    int t    = rest / KB;
    int n = t * 16 + (lane & 15);
    int k = kb * 32 + (lane >> 4) * 8 + (i & 7) + (i >> 3) * 16;
    float v = (k < K) ? W[(size_t)k * Nn + n] : 0.f;
    Wo[o] = (_Float16)v;
}

// ---------------------------------------------------------------------
// Per-edge geometry: diff = x_src - x_dst, d = |diff|
// ---------------------------------------------------------------------
__global__ void egcl_geom(const float* __restrict__ coords,
                          const int* __restrict__ edges,
                          float* __restrict__ diff, float* __restrict__ dbuf) {
    int e = blockIdx.x * blockDim.x + threadIdx.x;
    if (e >= NEDGE) return;
    int s = edges[2 * e], t = edges[2 * e + 1];
    float dx = coords[3 * s + 0] - coords[3 * t + 0];
    float dy = coords[3 * s + 1] - coords[3 * t + 1];
    float dz = coords[3 * s + 2] - coords[3 * t + 2];
    diff[3 * e + 0] = dx; diff[3 * e + 1] = dy; diff[3 * e + 2] = dz;
    dbuf[e] = sqrtf(dx * dx + dy * dy + dz * dz);
}

// ---------------------------------------------------------------------
// Build mlp_in f16 [NEDGE, KE]: [h_src | h_dst | d^2 | edge_attr | 0pad]
// ---------------------------------------------------------------------
__global__ void egcl_build_edge_in(const float* __restrict__ features,
                                   const float* __restrict__ edge_attr,
                                   const int* __restrict__ edges,
                                   const float* __restrict__ dbuf,
                                   _Float16* __restrict__ mlpIn) {
    size_t i = (size_t)blockIdx.x * blockDim.x + threadIdx.x;
    if (i >= (size_t)NEDGE * KE) return;
    int e = (int)(i / KE);
    int j = (int)(i % KE);
    float v;
    if (j < FDIM) {
        v = features[(size_t)edges[2 * e] * FDIM + j];
    } else if (j < 2 * FDIM) {
        v = features[(size_t)edges[2 * e + 1] * FDIM + (j - FDIM)];
    } else if (j == 2 * FDIM) {
        float d = dbuf[e]; v = d * d;
    } else if (j < 2 * FDIM + 1 + EADIM) {
        v = edge_attr[(size_t)e * EADIM + (j - (2 * FDIM + 1))];
    } else {
        v = 0.f;
    }
    mlpIn[i] = (_Float16)v;
}

// ---------------------------------------------------------------------
// Init: copy coords into d_out coord section, zero agg accumulator.
// ---------------------------------------------------------------------
__global__ void egcl_init(const float* __restrict__ coords,
                          float* __restrict__ coordsOut,
                          float* __restrict__ agg) {
    int i = blockIdx.x * blockDim.x + threadIdx.x;
    if (i < NNODE * 3) coordsOut[i] = coords[i];
    if (i < NNODE * HDIM) agg[i] = 0.f;
}

// ---------------------------------------------------------------------
// Fragment loader (ISA 16-bit WMMA A layout, wave32)
// ---------------------------------------------------------------------
__device__ __forceinline__ v16h egcl_load_a(const _Float16* __restrict__ arow,
                                            int kb, int kh) {
    const _Float16* ap = arow + kb * 32 + kh * 8;
    v8h lo = *(const v8h*)(ap);
    v8h hi = *(const v8h*)(ap + 16);
    v16h a;
#pragma unroll
    for (int q = 0; q < 8; ++q) { a[q] = lo[q]; a[8 + q] = hi[q]; }
    return a;
}

// ---------------------------------------------------------------------
// WMMA GEMM: Out_f16[M,Nn] = act(A_f16[M,K] @ Bfrag_f16[K,Nn] + bias)
// Register-blocked: one wave computes a 64x32 output tile
// (4 m-tiles x 2 n-tiles = 8 f32 accumulators, 8 WMMAs per k-step).
// Per k-step traffic: 4KB A + 2KB B for 128 KFLOP -> 21.3 FLOP/B from L2.
// VGPR budget: 64 acc + 32 A + 16 B + pipelining headroom < 256 (no spills).
// Requires M % 64 == 0, Nn % 32 == 0, K % 32 == 0.
// ---------------------------------------------------------------------
template <int ACT>
__global__ void egcl_gemm_f16(const _Float16* __restrict__ A,
                              const _Float16* __restrict__ Bp,
                              const float* __restrict__ bias,
                              _Float16* __restrict__ Out,
                              int M, int Nn, int K) {
    int wave = blockIdx.x * (blockDim.x >> 5) + (threadIdx.x >> 5);
    int lane = threadIdx.x & 31;
    int ngrp = Nn >> 5;                       // 32-col groups
    int mgrp = M >> 6;                        // 64-row groups
    if (wave >= mgrp * ngrp) return;
    int mg = wave / ngrp, ng = wave % ngrp;
    int KB = K >> 5;
    int kh = lane >> 4;
    int ml = lane & 15;

    const _Float16* arow[4];
#pragma unroll
    for (int r = 0; r < 4; ++r)
        arow[r] = A + (size_t)(mg * 64 + r * 16 + ml) * K;

    const v16h* bbase[2];
#pragma unroll
    for (int j = 0; j < 2; ++j)
        bbase[j] = (const v16h*)Bp + ((size_t)(ng * 2 + j) * KB) * 32 + lane;

    v8f acc[4][2] = {};
    for (int kb = 0; kb < KB; ++kb) {
        v16h a[4];
#pragma unroll
        for (int r = 0; r < 4; ++r) a[r] = egcl_load_a(arow[r], kb, kh);
#pragma unroll
        for (int j = 0; j < 2; ++j) {
            v16h b = bbase[j][(size_t)kb * 32];
#pragma unroll
            for (int r = 0; r < 4; ++r)
                acc[r][j] = __builtin_amdgcn_wmma_f32_16x16x32_f16(
                    false, a[r], false, b, (short)0, acc[r][j], false, false);
        }
    }

#pragma unroll
    for (int j = 0; j < 2; ++j) {
        int n = ng * 32 + j * 16 + ml;
        float bv = bias[n];
#pragma unroll
        for (int r = 0; r < 4; ++r) {
            int mbase = mg * 64 + r * 16 + kh * 8;
#pragma unroll
            for (int v = 0; v < 8; ++v) {
                float x = acc[r][j][v] + bv;
                if (ACT == 1) x = silu_f(x);
                Out[(size_t)(mbase + v) * Nn + n] = (_Float16)x;
            }
        }
    }
}

// Same GEMM, but f32 output with residual add (final node_mlp layer).
__global__ void egcl_gemm_f16_res_f32(const _Float16* __restrict__ A,
                                      const _Float16* __restrict__ Bp,
                                      const float* __restrict__ bias,
                                      const float* __restrict__ resid,
                                      float* __restrict__ Out,
                                      int M, int Nn, int K) {
    int wave = blockIdx.x * (blockDim.x >> 5) + (threadIdx.x >> 5);
    int lane = threadIdx.x & 31;
    int ngrp = Nn >> 5;
    int mgrp = M >> 6;
    if (wave >= mgrp * ngrp) return;
    int mg = wave / ngrp, ng = wave % ngrp;
    int KB = K >> 5;
    int kh = lane >> 4;
    int ml = lane & 15;

    const _Float16* arow[4];
#pragma unroll
    for (int r = 0; r < 4; ++r)
        arow[r] = A + (size_t)(mg * 64 + r * 16 + ml) * K;

    const v16h* bbase[2];
#pragma unroll
    for (int j = 0; j < 2; ++j)
        bbase[j] = (const v16h*)Bp + ((size_t)(ng * 2 + j) * KB) * 32 + lane;

    v8f acc[4][2] = {};
    for (int kb = 0; kb < KB; ++kb) {
        v16h a[4];
#pragma unroll
        for (int r = 0; r < 4; ++r) a[r] = egcl_load_a(arow[r], kb, kh);
#pragma unroll
        for (int j = 0; j < 2; ++j) {
            v16h b = bbase[j][(size_t)kb * 32];
#pragma unroll
            for (int r = 0; r < 4; ++r)
                acc[r][j] = __builtin_amdgcn_wmma_f32_16x16x32_f16(
                    false, a[r], false, b, (short)0, acc[r][j], false, false);
        }
    }

#pragma unroll
    for (int j = 0; j < 2; ++j) {
        int n = ng * 32 + j * 16 + ml;
        float bv = bias[n];
#pragma unroll
        for (int r = 0; r < 4; ++r) {
            int mbase = mg * 64 + r * 16 + kh * 8;
#pragma unroll
            for (int v = 0; v < 8; ++v) {
                size_t idx = (size_t)(mbase + v) * Nn + n;
                Out[idx] = resid[idx] + acc[r][j][v] + bv;
            }
        }
    }
}

// ---------------------------------------------------------------------
// Per-edge epilogue: gate = sigmoid(phi_e . Wi + bi), phi_x = xx2 . Wx3 + bx3
// scatter: agg[src] += gate*phi_e ; coordsOut[src] += (phi_x/(d+1)) * diff
// One wave per edge; wave32 shuffle reduction; f32 global atomics.
// ---------------------------------------------------------------------
__global__ void egcl_edge_scatter(const _Float16* __restrict__ phiE,
                                  const _Float16* __restrict__ xx2,
                                  const float* __restrict__ Wi, const float* __restrict__ bi,
                                  const float* __restrict__ Wx3, const float* __restrict__ bx3,
                                  const float* __restrict__ dbuf,
                                  const float* __restrict__ diff,
                                  const int* __restrict__ edges,
                                  float* __restrict__ agg,
                                  float* __restrict__ coordsOut) {
    int e = blockIdx.x * (blockDim.x >> 5) + (threadIdx.x >> 5);
    int lane = threadIdx.x & 31;
    if (e >= NEDGE) return;
    const _Float16* pe = phiE + (size_t)e * HDIM;
    const _Float16* px = xx2 + (size_t)e * HDIM;
    float sg = 0.f, sx = 0.f;
#pragma unroll
    for (int i = lane; i < HDIM; i += 32) {
        sg += (float)pe[i] * Wi[i];
        sx += (float)px[i] * Wx3[i];
    }
#pragma unroll
    for (int off = 16; off > 0; off >>= 1) {
        sg += __shfl_xor(sg, off, 32);
        sx += __shfl_xor(sx, off, 32);
    }
    float gate = 1.f / (1.f + __expf(-(sg + bi[0])));
    float w = (sx + bx3[0]) / (dbuf[e] + 1.f);
    int src = edges[2 * e];
#pragma unroll
    for (int i = lane; i < HDIM; i += 32)
        atomicAdd(&agg[(size_t)src * HDIM + i], gate * (float)pe[i]);
    if (lane < 3)
        atomicAdd(&coordsOut[src * 3 + lane], w * diff[3 * e + lane]);
}

// ---------------------------------------------------------------------
// Build node_in f16 [NNODE, KNODE]: [features | agg | node_attr | 0pad]
// ---------------------------------------------------------------------
__global__ void egcl_build_node_in(const float* __restrict__ features,
                                   const float* __restrict__ agg,
                                   const float* __restrict__ node_attr,
                                   _Float16* __restrict__ nodeIn) {
    int i = blockIdx.x * blockDim.x + threadIdx.x;
    if (i >= NNODE * KNODE) return;
    int n = i / KNODE, j = i % KNODE;
    float v;
    if (j < FDIM)                 v = features[(size_t)n * FDIM + j];
    else if (j < FDIM + HDIM)     v = agg[(size_t)n * HDIM + (j - FDIM)];
    else if (j < FDIM + HDIM + NADIM) v = node_attr[(size_t)n * NADIM + (j - FDIM - HDIM)];
    else                          v = 0.f;
    nodeIn[i] = (_Float16)v;
}

// ---------------------------------------------------------------------
extern "C" void kernel_launch(void* const* d_in, const int* in_sizes, int n_in,
                              void* d_out, int out_size, void* d_ws, size_t ws_size,
                              hipStream_t stream) {
    const float* coords    = (const float*)d_in[0];
    const float* features  = (const float*)d_in[1];
    const int*   edges     = (const int*)  d_in[2];
    // d_in[3] = reduce matrix (unused: scatter by edges[:,0] instead)
    const float* node_attr = (const float*)d_in[4];
    const float* edge_attr = (const float*)d_in[5];
    const float* We1 = (const float*)d_in[6];  const float* be1 = (const float*)d_in[7];
    const float* We2 = (const float*)d_in[8];  const float* be2 = (const float*)d_in[9];
    const float* Wx1 = (const float*)d_in[10]; const float* bx1 = (const float*)d_in[11];
    const float* Wx2 = (const float*)d_in[12]; const float* bx2 = (const float*)d_in[13];
    const float* Wx3 = (const float*)d_in[14]; const float* bx3 = (const float*)d_in[15];
    const float* Wn1 = (const float*)d_in[16]; const float* bn1 = (const float*)d_in[17];
    const float* Wn2 = (const float*)d_in[18]; const float* bn2 = (const float*)d_in[19];
    const float* Wi  = (const float*)d_in[20]; const float* bi  = (const float*)d_in[21];

    float* coordsOut = (float*)d_out;                 // [NNODE,3]
    float* featOut   = (float*)d_out + NNODE * 3;     // [NNODE,FDIM]

    // ---- workspace layout (256B aligned bump allocator) ----
    char* ws = (char*)d_ws;
    size_t off = 0;
    auto alloc = [&](size_t bytes) { size_t r = off; off = (off + bytes + 255) & ~(size_t)255; return r; };
    size_t oMlpIn = alloc((size_t)NEDGE * KE * 2);        // f16, also reused for Xx2
    size_t oX1    = alloc((size_t)NEDGE * HDIM * 2);      // f16
    size_t oXx1   = alloc((size_t)NEDGE * HDIM * 2);      // f16
    size_t oPhiE  = alloc((size_t)NEDGE * HDIM * 2);      // f16
    size_t oDiff  = alloc((size_t)NEDGE * 3 * 4);         // f32
    size_t oD     = alloc((size_t)NEDGE * 4);             // f32
    size_t oAgg   = alloc((size_t)NNODE * HDIM * 4);      // f32
    size_t oNodeIn= alloc((size_t)NNODE * KNODE * 2);     // f16
    size_t oTn    = alloc((size_t)NNODE * HDIM * 2);      // f16
    size_t oWe1p  = alloc((size_t)KE * HDIM * 2);
    size_t oWx1p  = alloc((size_t)KE * HDIM * 2);
    size_t oWe2p  = alloc((size_t)HDIM * HDIM * 2);
    size_t oWx2p  = alloc((size_t)HDIM * HDIM * 2);
    size_t oWn1p  = alloc((size_t)KNODE * HDIM * 2);
    size_t oWn2p  = alloc((size_t)HDIM * FDIM * 2);
    (void)ws_size;

    _Float16* mlpIn = (_Float16*)(ws + oMlpIn);
    _Float16* X1    = (_Float16*)(ws + oX1);
    _Float16* Xx1   = (_Float16*)(ws + oXx1);
    _Float16* PhiE  = (_Float16*)(ws + oPhiE);
    _Float16* Xx2   = (_Float16*)(ws + oMlpIn);   // alias: mlp_in dead after layer-1 GEMMs
    float*    Diff  = (float*)(ws + oDiff);
    float*    Dbuf  = (float*)(ws + oD);
    float*    Agg   = (float*)(ws + oAgg);
    _Float16* NodeIn= (_Float16*)(ws + oNodeIn);
    _Float16* Tn    = (_Float16*)(ws + oTn);
    _Float16* We1p  = (_Float16*)(ws + oWe1p);
    _Float16* Wx1p  = (_Float16*)(ws + oWx1p);
    _Float16* We2p  = (_Float16*)(ws + oWe2p);
    _Float16* Wx2p  = (_Float16*)(ws + oWx2p);
    _Float16* Wn1p  = (_Float16*)(ws + oWn1p);
    _Float16* Wn2p  = (_Float16*)(ws + oWn2p);

    const int TB = 256;
    auto blocksFor = [](size_t n, int tb) { return (int)((n + tb - 1) / tb); };

    // 1) convert weights -> f16 fragment layout
    egcl_cvt_weight_frag<<<blocksFor((size_t)KE * HDIM, TB), TB, 0, stream>>>(We1, We1p, DIN, HDIM, KE);
    egcl_cvt_weight_frag<<<blocksFor((size_t)KE * HDIM, TB), TB, 0, stream>>>(Wx1, Wx1p, DIN, HDIM, KE);
    egcl_cvt_weight_frag<<<blocksFor((size_t)HDIM * HDIM, TB), TB, 0, stream>>>(We2, We2p, HDIM, HDIM, HDIM);
    egcl_cvt_weight_frag<<<blocksFor((size_t)HDIM * HDIM, TB), TB, 0, stream>>>(Wx2, Wx2p, HDIM, HDIM, HDIM);
    egcl_cvt_weight_frag<<<blocksFor((size_t)KNODE * HDIM, TB), TB, 0, stream>>>(Wn1, Wn1p, FDIM + HDIM + NADIM, HDIM, KNODE);
    egcl_cvt_weight_frag<<<blocksFor((size_t)HDIM * FDIM, TB), TB, 0, stream>>>(Wn2, Wn2p, HDIM, FDIM, HDIM);

    // 2) geometry + edge-MLP input, init accumulators
    egcl_geom<<<blocksFor(NEDGE, TB), TB, 0, stream>>>(coords, edges, Diff, Dbuf);
    egcl_build_edge_in<<<blocksFor((size_t)NEDGE * KE, TB), TB, 0, stream>>>(features, edge_attr, edges, Dbuf, mlpIn);
    egcl_init<<<blocksFor((size_t)NNODE * HDIM, TB), TB, 0, stream>>>(coords, coordsOut, Agg);

    // 3) edge GEMMs (one wave per 64x32 tile, 8 waves/block)
    auto gemmBlocks = [](int M, int Nn) { return ((M >> 6) * (Nn >> 5) + 7) / 8; };
    egcl_gemm_f16<1><<<gemmBlocks(NEDGE, HDIM), TB, 0, stream>>>(mlpIn, We1p, be1, X1, NEDGE, HDIM, KE);
    egcl_gemm_f16<1><<<gemmBlocks(NEDGE, HDIM), TB, 0, stream>>>(mlpIn, Wx1p, bx1, Xx1, NEDGE, HDIM, KE);
    egcl_gemm_f16<1><<<gemmBlocks(NEDGE, HDIM), TB, 0, stream>>>(X1, We2p, be2, PhiE, NEDGE, HDIM, HDIM);
    egcl_gemm_f16<1><<<gemmBlocks(NEDGE, HDIM), TB, 0, stream>>>(Xx1, Wx2p, bx2, Xx2, NEDGE, HDIM, HDIM);

    // 4) gate + scatter (one wave per edge)
    egcl_edge_scatter<<<(NEDGE + 7) / 8, TB, 0, stream>>>(PhiE, Xx2, Wi, bi, Wx3, bx3,
                                                          Dbuf, Diff, edges, Agg, coordsOut);

    // 5) node MLP
    egcl_build_node_in<<<blocksFor((size_t)NNODE * KNODE, TB), TB, 0, stream>>>(features, Agg, node_attr, NodeIn);
    egcl_gemm_f16<1><<<gemmBlocks(NNODE, HDIM), TB, 0, stream>>>(NodeIn, Wn1p, bn1, Tn, NNODE, HDIM, KNODE);
    egcl_gemm_f16_res_f32<<<gemmBlocks(NNODE, FDIM), TB, 0, stream>>>(Tn, Wn2p, bn2, features, featOut, NNODE, FDIM, HDIM);
}
